// GATRegressor_2327872274534
// MI455X (gfx1250) — matrix operand order repaired and checked by reading, not compiled, not run on hardware
//
#include <hip/hip_runtime.h>
#include <hip/hip_bf16.h>
#include <math.h>

typedef __attribute__((ext_vector_type(16))) _Float16 v16h;
typedef __attribute__((ext_vector_type(8)))  _Float16 v8h;
typedef __attribute__((ext_vector_type(4)))  _Float16 v4h;
typedef __attribute__((ext_vector_type(8)))  float    v8f;

#define HIDC 128            // channels per head
#define NEG_SLOPE 0.2f
#define TM 64
#define TN 64
#define TK 32
#define FRAG_STRIDE 40      // halves per lane fragment slot: 80B, 16B-aligned

// ---------------------------------------------------------------- fill
__global__ void gat_fill_f32(float* __restrict__ p, float v, size_t n) {
  size_t i = (size_t)blockIdx.x * blockDim.x + threadIdx.x;
  if (i < n) p[i] = v;
}
__global__ void gat_fill_u32(unsigned* __restrict__ p, unsigned v, size_t n) {
  size_t i = (size_t)blockIdx.x * blockDim.x + threadIdx.x;
  if (i < n) p[i] = v;
}

// ---------------------------------------------------------------- GEMM (WMMA f16->f32)
// C[nrows,M] = A[nrows,K] @ W[K,M], f32 in memory, f16 on the matrix pipes.
// Block = 128 threads = 4 waves; block tile 64x64; wave tile 16x64.
// Double-buffered LDS staged directly in WMMA fragment layout: one barrier
// per K-step, wide ds_load_b128 fragment fetch, b128 global loads.

// stage one 64x32 A tile + 32x64 B tile (K-step k0) into fragment-layout LDS
template<bool FULL>
__device__ __forceinline__ void gat_stage(const float* __restrict__ A,
                                          const float* __restrict__ W,
                                          int nrows, int K, int M,
                                          int row0, int col0, int k0, int tid,
                                          _Float16 (&Asf)[4][32][FRAG_STRIDE],
                                          _Float16 (&Bsf)[4][32][FRAG_STRIDE]) {
  // A: quad q covers 4 consecutive K at row r -> frag elems 2vi..2vi+3 (one b64 store)
#pragma unroll
  for (int t = 0; t < 4; ++t) {
    int p  = tid + t * 128;
    int r  = p >> 3, kq = (p & 7) * 4;
    int gr = row0 + r;
    float4 f;
    if (FULL) {
      f = *(const float4*)&A[(size_t)gr * K + k0 + kq];
    } else {
      int cr = (gr < nrows) ? gr : (nrows - 1);
      f = *(const float4*)&A[(size_t)cr * K + k0 + kq];
      if (gr >= nrows) { f.x = 0.f; f.y = 0.f; f.z = 0.f; f.w = 0.f; }
    }
    int w  = r >> 4, m = r & 15;
    int hh = ((kq & 15) >= 8) ? 1 : 0;
    int vi = ((kq >= 16) ? 4 : 0) + ((kq & 7) >> 1);   // even: {0,2,4,6}
    v4h q; q[0] = (_Float16)f.x; q[1] = (_Float16)f.y;
           q[2] = (_Float16)f.z; q[3] = (_Float16)f.w;
    *(v4h*)&Asf[w][hh * 16 + m][2 * vi] = q;
  }
  // B: quad covers 4 consecutive K rows at column cc -> frag elems i..i+3 (one b64 store)
#pragma unroll
  for (int t = 0; t < 4; ++t) {
    int p  = tid + t * 128;
    int r  = (p >> 6) * 4, cc = p & 63;
    const float* wp = &W[(size_t)(k0 + r) * M + col0 + cc];
    float f0 = wp[0], f1 = wp[(size_t)M], f2 = wp[(size_t)2 * M], f3 = wp[(size_t)3 * M];
    int hh = (r >> 3) & 1, i = (r & 7) + ((r >= 16) ? 8 : 0);   // i in {0,4,8,12}
    int j  = cc >> 4, nn = cc & 15;
    v4h q; q[0] = (_Float16)f0; q[1] = (_Float16)f1;
           q[2] = (_Float16)f2; q[3] = (_Float16)f3;
    *(v4h*)&Bsf[j][hh * 16 + nn][i] = q;
  }
}

template<bool FULL>
__device__ __forceinline__ void gat_gemm_body(const float* __restrict__ A,
                                              const float* __restrict__ W,
                                              float* __restrict__ C,
                                              int nrows, int K, int M,
                                              int row0, int col0,
                                              _Float16 (&AsfBuf)[2][4][32][FRAG_STRIDE],
                                              _Float16 (&BsfBuf)[2][4][32][FRAG_STRIDE]) {
  const int tid  = threadIdx.x;
  const int lane = tid & 31;
  const int wave = tid >> 5;
  const int lm   = lane & 15;
  const int half = lane >> 4;

  v8f acc[4] = {v8f{}, v8f{}, v8f{}, v8f{}};

  gat_stage<FULL>(A, W, nrows, K, M, row0, col0, 0, tid, AsfBuf[0], BsfBuf[0]);
  __syncthreads();

  const int nk = K / TK;
  for (int ki = 0; ki < nk; ++ki) {
    const int cur = ki & 1;
    if (ki + 1 < nk) {
      if (ki + 2 < nk) {  // prefetch tile ki+2 into L2 while staging ki+1
        int pr = row0 + (tid >> 1); if (pr >= nrows) pr = nrows - 1;
        __builtin_prefetch(&A[(size_t)pr * K + (ki + 2) * TK + (tid & 1) * 16], 0, 1);
        __builtin_prefetch(&W[(size_t)((ki + 2) * TK + (tid >> 2)) * M + col0 + (tid & 3) * 16], 0, 1);
      }
      gat_stage<FULL>(A, W, nrows, K, M, row0, col0, (ki + 1) * TK, tid,
                      AsfBuf[1 - cur], BsfBuf[1 - cur]);
    }
    // fragment fetch: 2x ds_load_b128 per operand
    const v8h a0 = *(const v8h*)&AsfBuf[cur][wave][lane][0];
    const v8h a1 = *(const v8h*)&AsfBuf[cur][wave][lane][8];
    const v16h afrag = __builtin_shufflevector(a0, a1, 0,1,2,3,4,5,6,7,8,9,10,11,12,13,14,15);
#pragma unroll
    for (int j = 0; j < 4; ++j) {
      const v8h b0 = *(const v8h*)&BsfBuf[cur][j][lane][0];
      const v8h b1 = *(const v8h*)&BsfBuf[cur][j][lane][8];
      const v16h bfrag = __builtin_shufflevector(b0, b1, 0,1,2,3,4,5,6,7,8,9,10,11,12,13,14,15);
      acc[j] = __builtin_amdgcn_wmma_f32_16x16x32_f16(
          false, afrag, false, bfrag, (short)0, acc[j], false, false);
    }
    __syncthreads();   // staged (ki+1) ready; buf[cur] reads done before overwrite
  }

  // store: C VGPR r -> row = r + 8*half, col = lane%16 (ISA 7.12.2)
#pragma unroll
  for (int j = 0; j < 4; ++j) {
#pragma unroll
    for (int r = 0; r < 8; ++r) {
      int gr = row0 + wave * 16 + r + 8 * half;
      if (FULL || gr < nrows) C[(size_t)gr * M + col0 + j * 16 + lm] = acc[j][r];
    }
  }
}

__global__ __launch_bounds__(128)
void gat_gemm_wmma(const float* __restrict__ A, const float* __restrict__ W,
                   float* __restrict__ C, int nrows, int K, int M) {
  __shared__ __attribute__((aligned(16))) _Float16 AsfBuf[2][4][32][FRAG_STRIDE];
  __shared__ __attribute__((aligned(16))) _Float16 BsfBuf[2][4][32][FRAG_STRIDE];
  const int row0 = blockIdx.x * TM;
  const int col0 = blockIdx.y * TN;
  if (row0 + TM <= nrows)
    gat_gemm_body<true >(A, W, C, nrows, K, M, row0, col0, AsfBuf, BsfBuf);
  else
    gat_gemm_body<false>(A, W, C, nrows, K, M, row0, col0, AsfBuf, BsfBuf);
}

// ---------------------------------------------------------------- alpha_src / alpha_dst
// one wave per (node, head): dot(h[n,h,:], a_src[h,:]) and a_dst
__global__ __launch_bounds__(128)
void gat_alpha(const float* __restrict__ h, const float* __restrict__ a_src,
               const float* __restrict__ a_dst, float* __restrict__ als,
               float* __restrict__ ald, int n, int H) {
  size_t wid = ((size_t)blockIdx.x * blockDim.x + threadIdx.x) >> 5;
  int lane = threadIdx.x & 31;
  size_t total = (size_t)n * H;
  if (wid >= total) return;
  int node = (int)(wid / H), hh = (int)(wid % H);
  const float* hp  = h + (size_t)node * H * HIDC + (size_t)hh * HIDC;
  const float* sp  = a_src + (size_t)hh * HIDC;
  const float* dp  = a_dst + (size_t)hh * HIDC;
  float s = 0.f, d = 0.f;
  for (int c = lane; c < HIDC; c += 32) { float v = hp[c]; s += v * sp[c]; d += v * dp[c]; }
  for (int o = 16; o > 0; o >>= 1) { s += __shfl_xor(s, o, 32); d += __shfl_xor(d, o, 32); }
  if (lane == 0) { als[wid] = s; ald[wid] = d; }
}

// ---------------------------------------------------------------- edge helpers
__device__ __forceinline__ int edge_src(const int* ei, int E, size_t e) {
  return (e < (size_t)E) ? ei[e] : (int)(e - E);
}
__device__ __forceinline__ int edge_dst(const int* ei, int E, size_t e) {
  return (e < (size_t)E) ? ei[(size_t)E + e] : (int)(e - E);
}
// order-preserving float->uint key for atomicMax over signed floats
__device__ __forceinline__ unsigned f32_key(float f) {
  unsigned u = __float_as_uint(f);
  return (u & 0x80000000u) ? ~u : (u | 0x80000000u);
}
__device__ __forceinline__ float key_f32(unsigned k) {
  unsigned u = (k & 0x80000000u) ? (k & 0x7FFFFFFFu) : ~k;
  return __uint_as_float(u);
}

// pass 1: leaky-relu score, segment max via atomicMax on encoded uint
__global__ void gat_edge_score_max(const int* __restrict__ ei, int E, int n, int H,
                                   const float* __restrict__ als, const float* __restrict__ ald,
                                   float* __restrict__ ebuf, unsigned* __restrict__ mkey) {
  size_t idx = (size_t)blockIdx.x * blockDim.x + threadIdx.x;
  size_t total = (size_t)(E + n) * H;
  if (idx >= total) return;
  size_t e = idx / H; int hh = (int)(idx % H);
  int s = edge_src(ei, E, e), d = edge_dst(ei, E, e);
  float v = als[(size_t)s * H + hh] + ald[(size_t)d * H + hh];
  v = (v > 0.f) ? v : NEG_SLOPE * v;
  ebuf[idx] = v;
  atomicMax(&mkey[(size_t)d * H + hh], f32_key(v));
}

// pass 2: exp(e - m) and segment sum of denominators
__global__ void gat_edge_exp_sum(const int* __restrict__ ei, int E, int n, int H,
                                 const unsigned* __restrict__ mkey,
                                 float* __restrict__ ebuf, float* __restrict__ denom) {
  size_t idx = (size_t)blockIdx.x * blockDim.x + threadIdx.x;
  size_t total = (size_t)(E + n) * H;
  if (idx >= total) return;
  size_t e = idx / H; int hh = (int)(idx % H);
  int d = edge_dst(ei, E, e);
  float m = key_f32(mkey[(size_t)d * H + hh]);
  if (!(m > -3.0e38f) || !(m < 3.0e38f)) m = 0.f;   // isfinite fallback
  float ex = __expf(ebuf[idx] - m);
  ebuf[idx] = ex;
  atomicAdd(&denom[(size_t)d * H + hh], ex);
}

// pass 3: out[dst,h,:] += h[src,h,:] * alpha  — one wave per (edge, head)
__global__ __launch_bounds__(256)
void gat_edge_aggregate(const int* __restrict__ ei, int E, int n, int H,
                        const float* __restrict__ hsrc, const float* __restrict__ ebuf,
                        const float* __restrict__ denom, float* __restrict__ outp) {
  size_t wid = ((size_t)blockIdx.x * blockDim.x + threadIdx.x) >> 5;
  int lane = threadIdx.x & 31;
  size_t total = (size_t)(E + n) * H;
  if (wid >= total) return;
  size_t e = wid / H; int hh = (int)(wid % H);
  int s = edge_src(ei, E, e), d = edge_dst(ei, E, e);
  float alpha = ebuf[wid] / (denom[(size_t)d * H + hh] + 1e-16f);
  const float* hp = hsrc + (size_t)s * H * HIDC + (size_t)hh * HIDC;
  float* op = outp + (size_t)d * H * HIDC + (size_t)hh * HIDC;
#pragma unroll
  for (int c = lane; c < HIDC; c += 32) atomicAdd(&op[c], hp[c] * alpha);
}

// ---------------------------------------------------------------- bias (+ optional relu); M power of 2
__global__ void gat_bias_act(float* __restrict__ x, const float* __restrict__ b,
                             size_t total, int Mmask, int do_relu) {
  size_t i = (size_t)blockIdx.x * blockDim.x + threadIdx.x;
  if (i >= total) return;
  float v = x[i] + b[i & (size_t)Mmask];
  x[i] = do_relu ? fmaxf(v, 0.f) : v;
}

// ---------------------------------------------------------------- global mean pool + fc
__global__ __launch_bounds__(128)
void gat_pool_accum(const float* __restrict__ h, const int* __restrict__ batch,
                    float* __restrict__ sums, float* __restrict__ cnt, int n) {
  size_t wid = ((size_t)blockIdx.x * blockDim.x + threadIdx.x) >> 5;
  int lane = threadIdx.x & 31;
  if (wid >= (size_t)n) return;
  int b = batch[wid];
  const float* hp = h + wid * HIDC;
#pragma unroll
  for (int c = lane; c < HIDC; c += 32) atomicAdd(&sums[(size_t)b * HIDC + c], hp[c]);
  if (lane == 0) atomicAdd(&cnt[b], 1.0f);
}

__global__ __launch_bounds__(128)
void gat_pool_fc(const float* __restrict__ sums, const float* __restrict__ cnt,
                 const float* __restrict__ fcw, const float* __restrict__ fcb,
                 float* __restrict__ out, int Bg) {
  size_t wid = ((size_t)blockIdx.x * blockDim.x + threadIdx.x) >> 5;
  int lane = threadIdx.x & 31;
  if (wid >= (size_t)Bg) return;
  float c = fmaxf(cnt[wid], 1.0f);
  float acc = 0.f;
  for (int j = lane; j < HIDC; j += 32) acc += (sums[wid * HIDC + j] / c) * fcw[j];
  for (int o = 16; o > 0; o >>= 1) acc += __shfl_xor(acc, o, 32);
  if (lane == 0) out[wid] = acc + fcb[0];
}

// ---------------------------------------------------------------- host-side layer driver
static void run_gat_layer(const float* x, const float* W, const float* asv, const float* adv,
                          const float* bias, float* hbuf, float* obuf,
                          float* als, float* ald, unsigned* mkey, float* denom, float* ebuf,
                          const int* ei, int N, int E, int K, int H, int do_relu,
                          hipStream_t st) {
  const int M = H * HIDC;
  dim3 gg((N + TM - 1) / TM, M / TN);
  gat_gemm_wmma<<<gg, 128, 0, st>>>(x, W, hbuf, N, K, M);

  size_t nh = (size_t)N * H;
  gat_alpha<<<(unsigned)((nh * 32 + 127) / 128), 128, 0, st>>>(hbuf, asv, adv, als, ald, N, H);

  // encode(-inf) = ~bits(-inf) = 0x007FFFFF
  gat_fill_u32<<<(unsigned)((nh + 255) / 256), 256, 0, st>>>(mkey, 0x007FFFFFu, nh);
  gat_fill_f32<<<(unsigned)((nh + 255) / 256), 256, 0, st>>>(denom, 0.f, nh);
  size_t nm = (size_t)N * M;
  gat_fill_f32<<<(unsigned)((nm + 255) / 256), 256, 0, st>>>(obuf, 0.f, nm);

  size_t tot = (size_t)(E + N) * H;
  gat_edge_score_max<<<(unsigned)((tot + 255) / 256), 256, 0, st>>>(ei, E, N, H, als, ald, ebuf, mkey);
  gat_edge_exp_sum<<<(unsigned)((tot + 255) / 256), 256, 0, st>>>(ei, E, N, H, mkey, ebuf, denom);
  gat_edge_aggregate<<<(unsigned)((tot * 32 + 255) / 256), 256, 0, st>>>(ei, E, N, H, hbuf, ebuf, denom, obuf);

  gat_bias_act<<<(unsigned)((nm + 255) / 256), 256, 0, st>>>(obuf, bias, nm, M - 1, do_relu);
}

extern "C" void kernel_launch(void* const* d_in, const int* in_sizes, int n_in,
                              void* d_out, int out_size, void* d_ws, size_t ws_size,
                              hipStream_t stream) {
  const float* x    = (const float*)d_in[0];
  const int*   ei   = (const int*)  d_in[1];
  const int*   batch= (const int*)  d_in[2];
  const float* W1   = (const float*)d_in[3];
  const float* as1  = (const float*)d_in[4];
  const float* ad1  = (const float*)d_in[5];
  const float* b1   = (const float*)d_in[6];
  const float* W2   = (const float*)d_in[7];
  const float* as2  = (const float*)d_in[8];
  const float* ad2  = (const float*)d_in[9];
  const float* b2   = (const float*)d_in[10];
  const float* W3   = (const float*)d_in[11];
  const float* as3  = (const float*)d_in[12];
  const float* ad3  = (const float*)d_in[13];
  const float* b3   = (const float*)d_in[14];
  const float* fcw  = (const float*)d_in[15];
  const float* fcb  = (const float*)d_in[16];

  const int N  = in_sizes[2];        // batch vector length = #nodes
  const int E  = in_sizes[1] / 2;    // edge_index is [2,E]
  const int Bg = out_size;           // #graphs
  const int FIN = in_sizes[0] / N;   // 128
  const int H1 = 4;
  float* out = (float*)d_out;

  // workspace carve-up (f32 elements)
  size_t nA = (size_t)N * (H1 * HIDC);          // N x 512
  float*    bufA  = (float*)d_ws;
  float*    bufB  = bufA + nA;
  float*    als   = bufB + nA;
  float*    ald   = als + (size_t)N * H1;
  unsigned* mkey  = (unsigned*)(ald + (size_t)N * H1);
  float*    denom = (float*)(mkey + (size_t)N * H1);
  float*    ebuf  = denom + (size_t)N * H1;
  float*    sums  = ebuf + (size_t)(E + N) * H1;
  float*    cnt   = sums + (size_t)Bg * HIDC;

  // layer 1: x[N,128] -> bufA (h1) -> aggregate -> bufB (+b1, relu)
  run_gat_layer(x,    W1, as1, ad1, b1, bufA, bufB, als, ald, mkey, denom, ebuf,
                ei, N, E, FIN,      H1, 1, stream);
  // layer 2: bufB[N,512] -> bufA (h2) -> aggregate -> bufB (+b2, relu)
  run_gat_layer(bufB, W2, as2, ad2, b2, bufA, bufB, als, ald, mkey, denom, ebuf,
                ei, N, E, H1 * HIDC, H1, 1, stream);
  // layer 3: bufB[N,512] -> bufA (h3, 1 head) -> aggregate -> bufB (+b3)
  run_gat_layer(bufB, W3, as3, ad3, b3, bufA, bufB, als, ald, mkey, denom, ebuf,
                ei, N, E, H1 * HIDC, 1, 0, stream);

  // global mean pool + fc
  size_t ps = (size_t)Bg * HIDC;
  gat_fill_f32<<<(unsigned)((ps + 255) / 256), 256, 0, stream>>>(sums, 0.f, ps);
  gat_fill_f32<<<(unsigned)((Bg + 255) / 256), 256, 0, stream>>>(cnt, 0.f, (size_t)Bg);
  gat_pool_accum<<<(unsigned)(((size_t)N * 32 + 127) / 128), 128, 0, stream>>>(bufB, batch, sums, cnt, N);
  gat_pool_fc<<<(unsigned)(((size_t)Bg * 32 + 127) / 128), 128, 0, stream>>>(sums, cnt, fcw, fcb, out, Bg);
}
